// Block_5583457485503
// MI455X (gfx1250) — matrix operand order repaired and checked by reading, not compiled, not run on hardware
//
#include <hip/hip_runtime.h>
#include <hip/hip_bf16.h>

// ---------------------------------------------------------------------------
// Transformer block (causal MHSA + FFN), MI455X / gfx1250, wave32 + WMMA bf16.
//
// Shapes: B=8192, T=32, C=128, H=4, D=32, FF=512.
// One wave per batch element, 8 waves (256 threads) per workgroup.
// - x tile staged by the Tensor Data Mover (tensor_load_to_lds + TENSORcnt),
//   then converted f32 -> bf16 in LDS.
// - All GEMMs use v_wmma_f32_16x16x32_bf16 (f32 accumulate); A fragments use
//   the interleaved K layout, B fragments the same interleaved K layout.
// ---------------------------------------------------------------------------

#define NB     8192
#define TT     32
#define CC     128
#define NH     4
#define HD     32
#define FF     512
#define BPB    8        // batch elements per block (one per wave)

typedef __bf16 bf16;
typedef __attribute__((ext_vector_type(16))) __bf16    v16bf;
typedef __attribute__((ext_vector_type(8)))  __bf16    v8bf;
typedef __attribute__((ext_vector_type(8)))  float     v8f;
typedef __attribute__((ext_vector_type(4)))  float     v4f;
typedef __attribute__((ext_vector_type(4)))  unsigned  u32x4;
typedef __attribute__((ext_vector_type(8)))  unsigned  u32x8;

// ---- LDS layout (bytes). Phase-1 and phase-2 regions overlap in a union. ----
#define XB_OFF    0                    // x' bf16: 8 waves * 32*128*2 = 64 KB
#define U_OFF     65536
#define XSTG_OFF  U_OFF                // phase0: TDM f32 x staging: 8*16KB = 128 KB
#define QKVT_OFF  U_OFF                // wq/wk/wv transposed bf16: 3*4*32*128*2 = 96 KB
#define SCR_OFF   (U_OFF + 98304)      // per-wave q/k/vT scratch: 8 * 3*32*32*2 = 48 KB
#define AO_OFF    (SCR_OFF + 49152)    // attention out bf16: 8 * 32*128*2 = 64 KB
#define W1T_OFF   U_OFF                // phase2: w1 chunk^T  128*128*2 = 32 KB
#define W2T_OFF   (U_OFF + 32768)      // phase2: w2 chunk^T  32 KB
#define H_OFF     (U_OFF + 65536)      // phase2: h chunk: 8 * 32*128*2 = 64 KB
#define SMEM_BYTES 278528              // 272 KB  (<= 320 KB per WGP)

// A fragment, 16x32 bf16 from row-major M x K (CDNA5 16-bit A layout):
//   lanes 0-15 : row r, K = {k0..k0+7, k0+16..k0+23}
//   lanes 16-31: row r, K = {k0+8..k0+15, k0+24..k0+31}
__device__ __forceinline__ v16bf ldfragA(const bf16* base, int r0, int k0,
                                         int stride, int lane) {
  const int r  = lane & 15;
  const int hf = lane >> 4;
  const bf16* p = base + (r0 + r) * stride + k0 + hf * 8;
  v8bf lo = *(const v8bf*)p;          // ds_load_b128
  v8bf hi = *(const v8bf*)(p + 16);
  v16bf f;
#pragma unroll
  for (int i = 0; i < 8; ++i) { f[i] = lo[i]; f[i + 8] = hi[i]; }
  return f;
}

// B fragment, 32x16 bf16 (K x N) from BT[n][k] row-major (same K order as A):
//   lanes 0-15 : col n, K = {k0..k0+7, k0+16..k0+23}
//   lanes 16-31: col n, K = {k0+8..k0+15, k0+24..k0+31}
__device__ __forceinline__ v16bf ldfragB(const bf16* baseT, int n0, int k0,
                                         int stride, int lane) {
  const int r  = lane & 15;
  const int hf = lane >> 4;
  const bf16* p = baseT + (n0 + r) * stride + k0 + hf * 8;
  v8bf lo = *(const v8bf*)p;          // ds_load_b128
  v8bf hi = *(const v8bf*)(p + 16);
  v16bf f;
#pragma unroll
  for (int i = 0; i < 8; ++i) { f[i] = lo[i]; f[i + 8] = hi[i]; }
  return f;
}

__device__ __forceinline__ v8f wmma_bf16(v16bf a, v16bf b, v8f c) {
  // (neg_a, A, neg_b, B, c_mod, C, reuse_a, reuse_b)
  return __builtin_amdgcn_wmma_f32_16x16x32_bf16(false, a, false, b,
                                                 (short)0, c, false, false);
}

extern "C" __global__ void __launch_bounds__(256)
attn_ffn_block(const float* __restrict__ x,
               const float* __restrict__ wq, const float* __restrict__ wk,
               const float* __restrict__ wv,
               const float* __restrict__ w1, const float* __restrict__ b1,
               const float* __restrict__ w2, const float* __restrict__ b2,
               float* __restrict__ out) {
  extern __shared__ char smem[];
  bf16* xb   = (bf16*)(smem + XB_OFF);
  bf16* qkvT = (bf16*)(smem + QKVT_OFF);
  bf16* scr  = (bf16*)(smem + SCR_OFF);
  bf16* ao   = (bf16*)(smem + AO_OFF);
  bf16* w1T  = (bf16*)(smem + W1T_OFF);
  bf16* w2T  = (bf16*)(smem + W2T_OFF);
  bf16* hbuf = (bf16*)(smem + H_OFF);

  const int tid  = threadIdx.x;
  const int wave = tid >> 5;
  const int lane = tid & 31;
  const int bidx = blockIdx.x * BPB + wave;
  const int r    = lane & 15;
  const int hf   = lane >> 4;

  // Warm L2 with the (tiny, reused-by-every-WG) weight tensors.
  if (tid < 8) {
    __builtin_prefetch(w1 + tid * 8192, 0, 1);   // global_prefetch_b8
    __builtin_prefetch(w2 + tid * 8192, 0, 1);
  }

  // ---- phase 0: TDM-DMA this wave's 16 KB f32 x tile into LDS staging ----
  float* xstage = (float*)(smem + XSTG_OFF) + wave * (TT * CC);
  {
    unsigned long long ga =
        (unsigned long long)(const void*)(x + (size_t)bidx * TT * CC);
    unsigned lds_off = __builtin_amdgcn_readfirstlane(
        (unsigned)(unsigned long long)(uintptr_t)xstage);
    unsigned g_lo = __builtin_amdgcn_readfirstlane((unsigned)ga);
    unsigned g_hi = __builtin_amdgcn_readfirstlane((unsigned)(ga >> 32));

    // D# group 0: count=1 | lds_addr | global_addr[56:0] | type=2
    u32x4 g0;
    g0[0] = 1u;                                   // count=1, user mode
    g0[1] = lds_off;                              // lds_addr [63:32]
    g0[2] = g_lo;                                 // global_addr [95:64]
    g0[3] = 0x80000000u | (g_hi & 0x01FFFFFFu);   // ga[56:32] | type=2

    // D# group 1: data_size=4B, 1-D tile of 4096 elements (16 KB linear copy)
    u32x8 g1;
    g1[0] = 0x00020000u;          // workgroup_mask=0, data_size=2 (4 bytes)
    g1[1] = (4096u & 0xFFFFu) << 16;  // tensor_dim0[15:0]  (bits 63:48)
    g1[2] = 1u << 16;             // tensor_dim0[31:16]=0, tensor_dim1=1
    g1[3] = 4096u << 16;          // tensor_dim1 hi=0, tile_dim0=4096
    g1[4] = 0u;                   // tile_dim1=0 (unused), tile_dim2=0
    g1[5] = 4096u;                // tensor_dim0_stride lo
    g1[6] = 0u;                   // stride0 hi, tensor_dim1_stride lo
    g1[7] = 0u;                   // tensor_dim1_stride hi

    // EXEC is ignored for tensor ops: one DMA per wave. Tracked by TENSORcnt.
    asm volatile("tensor_load_to_lds %0, %1" :: "s"(g0), "s"(g1) : "memory");
    __builtin_amdgcn_s_wait_tensorcnt(0);
  }

  // convert staged f32 -> bf16 working tile
  bf16* xbw = xb + wave * (TT * CC);
#pragma unroll 4
  for (int i = lane; i < TT * CC / 4; i += 32) {
    v4f v = *((const v4f*)xstage + i);   // ds_load_b128
    bf16* dst = xbw + i * 4;
    dst[0] = (bf16)v[0]; dst[1] = (bf16)v[1];
    dst[2] = (bf16)v[2]; dst[3] = (bf16)v[3];
  }
  __syncthreads();   // xstage region is about to be overwritten by qkvT

  // ---- stage wq/wk/wv transposed: qkvT[((m*NH+h)*HD + d)*CC + c] ----
  {
    const float* srcs[3] = {wq, wk, wv};
    for (int m = 0; m < 3; ++m) {
      const float* s = srcs[m];
      for (int idx = tid; idx < NH * CC * HD; idx += 256) {
        int h   = idx >> 12;        // / (128*32)
        int rem = idx & 4095;
        int c   = rem >> 5;
        int d   = rem & 31;
        qkvT[((m * NH + h) * HD + d) * CC + c] = (bf16)s[idx];
      }
    }
  }
  __syncthreads();

  // =========================== attention ===========================
  bf16* qs  = scr + wave * 3072;   // 32x32 row-major [t][d] -> A of scores
  bf16* ks  = qs + 1024;           // 32x32 row-major [s][d] -> BT of scores
  bf16* vsT = ks + 1024;           // 32x32 transposed [d][s] -> BT of att@V
  bf16* aow = ao + wave * (TT * CC);

  for (int h = 0; h < NH; ++h) {
    // Q/K/V = x(32x128) @ w_h(128x32) -> 16 WMMA each
#pragma unroll
    for (int m = 0; m < 3; ++m) {
      const bf16* wT = qkvT + (m * NH + h) * HD * CC;
#pragma unroll
      for (int Mt = 0; Mt < 2; ++Mt)
#pragma unroll
        for (int Nt = 0; Nt < 2; ++Nt) {
          v8f acc = {};
#pragma unroll
          for (int kk = 0; kk < 4; ++kk) {
            v16bf a = ldfragA(xbw, Mt * 16, kk * 32, CC, lane);
            v16bf b = ldfragB(wT,  Nt * 16, kk * 32, CC, lane);
            acc = wmma_bf16(a, b, acc);
          }
#pragma unroll
          for (int rv = 0; rv < 8; ++rv) {
            int t = Mt * 16 + rv + hf * 8;   // C-frag: row = vgpr(+8*half)
            int d = Nt * 16 + r;             //         col = lane%16
            if (m == 0)      qs[t * HD + d]  = (bf16)acc[rv];
            else if (m == 1) ks[t * HD + d]  = (bf16)acc[rv];
            else             vsT[d * TT + t] = (bf16)acc[rv];
          }
        }
    }
    __syncthreads();

    // scores = Q @ K^T (K=32 fits one WMMA), scale + causal softmax
    v8f sc[2][2];
#pragma unroll
    for (int Mt = 0; Mt < 2; ++Mt)
#pragma unroll
      for (int Nt = 0; Nt < 2; ++Nt) {
        v8f z = {};
        v16bf a = ldfragA(qs, Mt * 16, 0, HD, lane);
        v16bf b = ldfragB(ks, Nt * 16, 0, HD, lane);
        sc[Mt][Nt] = wmma_bf16(a, b, z);
      }
    const float scale = 0.17677669529663687f;   // 32^-0.5
#pragma unroll
    for (int Mt = 0; Mt < 2; ++Mt) {
#pragma unroll
      for (int rv = 0; rv < 8; ++rv) {
        int t = Mt * 16 + rv + hf * 8;
        float a0 = sc[Mt][0][rv] * scale;       // s = r
        float a1 = sc[Mt][1][rv] * scale;       // s = 16 + r
        bool v0 = r <= t, v1 = (16 + r) <= t;
        a0 = v0 ? a0 : -1e30f;
        a1 = v1 ? a1 : -1e30f;
        float mx = fmaxf(a0, a1);
#pragma unroll
        for (int s = 1; s < 16; s <<= 1) mx = fmaxf(mx, __shfl_xor(mx, s, 32));
        float e0 = v0 ? __expf(a0 - mx) : 0.f;
        float e1 = v1 ? __expf(a1 - mx) : 0.f;
        float sum = e0 + e1;
#pragma unroll
        for (int s = 1; s < 16; s <<= 1) sum += __shfl_xor(sum, s, 32);
        float inv = 1.0f / sum;
        sc[Mt][0][rv] = e0 * inv;
        sc[Mt][1][rv] = e1 * inv;
      }
    }
    __syncthreads();
    // att (bf16) -> reuse qs as [t][s]
#pragma unroll
    for (int Mt = 0; Mt < 2; ++Mt)
#pragma unroll
      for (int Nt = 0; Nt < 2; ++Nt)
#pragma unroll
        for (int rv = 0; rv < 8; ++rv) {
          int t = Mt * 16 + rv + hf * 8;
          int s = Nt * 16 + r;
          qs[t * TT + s] = (bf16)sc[Mt][Nt][rv];
        }
    __syncthreads();

    // out_h = att(32x32) @ V(32x32)
#pragma unroll
    for (int Mt = 0; Mt < 2; ++Mt)
#pragma unroll
      for (int Nt = 0; Nt < 2; ++Nt) {
        v8f acc = {};
        v16bf a = ldfragA(qs,  Mt * 16, 0, TT, lane);
        v16bf b = ldfragB(vsT, Nt * 16, 0, TT, lane);
        acc = wmma_bf16(a, b, acc);
#pragma unroll
        for (int rv = 0; rv < 8; ++rv) {
          int t  = Mt * 16 + rv + hf * 8;
          int dl = Nt * 16 + r;
          aow[t * CC + h * HD + dl] = (bf16)acc[rv];
        }
      }
    __syncthreads();
  }

  // x' = x + attn_out  (kept bf16 in xb; also the final residual source)
  for (int i = lane; i < TT * CC; i += 32)
    xbw[i] = (bf16)((float)xbw[i] + (float)aow[i]);
  __syncthreads();   // also guards qkvT region before FFN weights overwrite it

  // ============================== FFN ==============================
  // h = relu(x'@w1+b1) streamed in 4 chunks of 128 cols; out += h_j@w2_j.
  v8f oacc[16];
#pragma unroll
  for (int i = 0; i < 16; ++i) oacc[i] = (v8f){};

  bf16* hw = hbuf + wave * (TT * CC);   // this wave's 32x128 h chunk

  for (int j = 0; j < 4; ++j) {
    __syncthreads();   // previous chunk's weight reads complete
    // w1 chunk transposed: w1T[n][c] = w1[c][j*128+n]   (coalesced reads)
    for (int idx = tid; idx < CC * 128; idx += 256) {
      int c = idx >> 7, n = idx & 127;
      w1T[n * CC + c] = (bf16)w1[c * FF + j * 128 + n];
    }
    // w2 chunk transposed: w2T[n][kc] = w2[j*128+kc][n]
    for (int idx = tid; idx < CC * 128; idx += 256) {
      int kc = idx >> 7, n = idx & 127;
      w2T[n * CC + kc] = (bf16)w2[(j * 128 + kc) * CC + n];
    }
    __syncthreads();

    // h_j = relu(x' @ w1_j + b1_j)   (2x8 tiles, K=128)
#pragma unroll
    for (int Nt = 0; Nt < 8; ++Nt) {
      float bias = b1[j * 128 + Nt * 16 + r];
#pragma unroll
      for (int Mt = 0; Mt < 2; ++Mt) {
        v8f acc = {};
#pragma unroll
        for (int kk = 0; kk < 4; ++kk) {
          v16bf a = ldfragA(xbw, Mt * 16, kk * 32, CC, lane);
          v16bf b = ldfragB(w1T, Nt * 16, kk * 32, CC, lane);
          acc = wmma_bf16(a, b, acc);
        }
#pragma unroll
        for (int rv = 0; rv < 8; ++rv) {
          int t = Mt * 16 + rv + hf * 8;
          hw[t * CC + Nt * 16 + r] = (bf16)fmaxf(acc[rv] + bias, 0.f);
        }
      }
    }
    __syncthreads();

    // out += h_j @ w2_j   (persistent f32 accumulators)
#pragma unroll
    for (int Mt = 0; Mt < 2; ++Mt)
#pragma unroll
      for (int Nt = 0; Nt < 8; ++Nt) {
        v8f acc = oacc[Mt * 8 + Nt];
#pragma unroll
        for (int kk = 0; kk < 4; ++kk) {
          v16bf a = ldfragA(hw,  Mt * 16, kk * 32, CC, lane);
          v16bf b = ldfragB(w2T, Nt * 16, kk * 32, CC, lane);
          acc = wmma_bf16(a, b, acc);
        }
        oacc[Mt * 8 + Nt] = acc;
      }
  }

  // ---- final: out = x' + ffn_out + b2  (f32, coalesced stores) ----
  float* og = out + (size_t)bidx * TT * CC;
#pragma unroll
  for (int Mt = 0; Mt < 2; ++Mt)
#pragma unroll
    for (int Nt = 0; Nt < 8; ++Nt) {
      int f = Nt * 16 + r;
      float bias = b2[f];
      v8f acc = oacc[Mt * 8 + Nt];
#pragma unroll
      for (int rv = 0; rv < 8; ++rv) {
        int t = Mt * 16 + rv + hf * 8;
        og[t * CC + f] = (float)xbw[t * CC + f] + acc[rv] + bias;
      }
    }
}

extern "C" void kernel_launch(void* const* d_in, const int* in_sizes, int n_in,
                              void* d_out, int out_size, void* d_ws, size_t ws_size,
                              hipStream_t stream) {
  (void)in_sizes; (void)n_in; (void)d_ws; (void)ws_size; (void)out_size;
  const float* x  = (const float*)d_in[0];
  const float* wq = (const float*)d_in[1];
  const float* wk = (const float*)d_in[2];
  const float* wv = (const float*)d_in[3];
  const float* w1 = (const float*)d_in[4];
  const float* b1 = (const float*)d_in[5];
  const float* w2 = (const float*)d_in[6];
  const float* b2 = (const float*)d_in[7];

  // 272 KB dynamic LDS per workgroup (gfx1250 WGP has 320 KB)
  static_assert(SMEM_BYTES <= 320 * 1024, "LDS budget");
  hipFuncSetAttribute((const void*)attn_ffn_block,
                      hipFuncAttributeMaxDynamicSharedMemorySize, SMEM_BYTES);

  attn_ffn_block<<<dim3(NB / BPB), dim3(256), SMEM_BYTES, stream>>>(
      x, wq, wk, wv, w1, b1, w2, b2, (float*)d_out);
}
